// GeometricStructure_47021301957081
// MI455X (gfx1250) — matrix operand order repaired and checked by reading, not compiled, not run on hardware
//
#include <hip/hip_runtime.h>
#include <cstdint>
#include <cstddef>

// ---------------------------------------------------------------------------
// GeometricStructure DAG on gfx1250 via v_wmma_f32_16x16x32_f16.
// DAG collapses: p2==p1, c2==c1, l3==l2. Pipeline of WMMA GEMM stages with
// f16 activations (f32 accumulate) staged through workspace (L2-resident).
// Double-buffered LDS A-tiles overlap staging with WMMA compute.
// ---------------------------------------------------------------------------

typedef __attribute__((ext_vector_type(16))) _Float16 v16h;
typedef __attribute__((ext_vector_type(8)))  float    v8f;

union Frag {
    v16h     h;
    uint32_t u[8];
    uint4    q[2];
};

static constexpr int B_   = 32768;
static constexpr int HR   = 132;   // real hidden width
static constexpr int HP   = 160;   // padded hidden width (5 x 32 K-tiles)
static constexpr int NBM  = B_ / 128;
static constexpr int SBUF = 128 * 17;   // one LDS A-tile buffer (dwords)

// ---------------------------------------------------------------------------
// Weight packing: dense f16 B-matrix fragment layout (16x16x32 WMMA).
// Fragment fi = nt*Kt + kt. Per lane: n = nt*16 + (lane&15);
// lanes 0-15 hold K = kt*32 + [0..15], lanes 16-31 hold K = kt*32 + [16..31];
// VGPR v holds the pair (2v, 2v+1) within that 16-wide K window (low half first).
// ---------------------------------------------------------------------------
__global__ void pack_w_kernel(const float* __restrict__ src,
                              const float* __restrict__ src2, int ld,
                              int Kreal, int Nreal, int Kt, int Nt,
                              uint16_t* __restrict__ dst, float scale) {
    int tid   = blockIdx.x * blockDim.x + threadIdx.x;
    int total = Kt * Nt * 32;
    if (tid >= total) return;
    int lane = tid & 31;
    int f    = tid >> 5;
    int kt   = f % Kt;
    int nt   = f / Kt;
    int n    = nt * 16 + (lane & 15);
    int kb   = kt * 32 + ((lane & 16) ? 16 : 0);
    uint32_t* d = (uint32_t*)dst + ((size_t)f * 32 + lane) * 8;
#pragma unroll
    for (int v = 0; v < 8; ++v) {
        int   k0 = kb + 2 * v;
        float a0 = 0.f, a1 = 0.f;
        if (n < Nreal) {
            if (k0 < Kreal) {
                a0 = src[(size_t)k0 * ld + n];
                if (src2) a0 += src2[(size_t)k0 * ld + n];
                a0 *= scale;
            }
            if (k0 + 1 < Kreal) {
                a1 = src[(size_t)(k0 + 1) * ld + n];
                if (src2) a1 += src2[(size_t)(k0 + 1) * ld + n];
                a1 *= scale;
            }
        }
        union { _Float16 h[2]; uint32_t u; } p;
        p.h[0] = (_Float16)a0;
        p.h[1] = (_Float16)a1;
        d[v] = p.u;
    }
}

__global__ void pack_b_kernel(const float* __restrict__ src, int Nreal, int Npad,
                              float* __restrict__ dst, float scale) {
    int i = blockIdx.x * blockDim.x + threadIdx.x;
    if (i < Npad) dst[i] = (i < Nreal) ? src[i] * scale : 0.f;
}

// ---------------------------------------------------------------------------
// GEMM: 8 waves per block; wave w owns rows [blockRow + 16w, +16) and WN
// 16-wide N tiles. A tile (128x32 f16) staged through double-buffered LDS,
// row stride padded to 17 dwords -> conflict-free ds gathers in the
// contiguous-K (k = 16h + s) A-frag layout. Staging of tile kt+1 overlaps WMMAs of tile kt.
// ---------------------------------------------------------------------------
template <int WN, int KT, bool SRCF32>
__device__ __forceinline__ void gemm_accum(const void* __restrict__ Aptr,
                                           const uint16_t* __restrict__ Wp,
                                           int ntBase, v8f* acc,
                                           uint32_t* sA, int blockRow) {
    constexpr int K = KT * 32;
    const int tid  = threadIdx.x;
    const int lane = tid & 31;
    const int aBase =
        (((tid >> 5) << 4) | (lane & 15)) * 17 + ((lane & 16) ? 8 : 0);
    const int srow = tid >> 1, shalf = tid & 1;
    const size_t rowOff = (size_t)(blockRow + srow) * K + shalf * 16;
    const int sdstOff = srow * 17 + shalf * 8;

    float4 sf[4];   // f32 staging regs
    uint4  sq[2];   // f16 staging regs

    auto issueLoads = [&](int kt) {
        if constexpr (SRCF32) {
            const float4* src = (const float4*)((const float*)Aptr + rowOff + kt * 32);
#pragma unroll
            for (int j = 0; j < 4; ++j) sf[j] = src[j];
        } else {
            const uint4* src = (const uint4*)((const uint16_t*)Aptr + rowOff + kt * 32);
            sq[0] = src[0];
            sq[1] = src[1];
        }
    };
    auto storeStage = [&](int kt) {
        uint32_t* d = sA + (kt & 1) * SBUF + sdstOff;
        if constexpr (SRCF32) {
#pragma unroll
            for (int j = 0; j < 4; ++j) {
                union { _Float16 h[2]; uint32_t u; } p0, p1;
                p0.h[0] = (_Float16)sf[j].x; p0.h[1] = (_Float16)sf[j].y;
                p1.h[0] = (_Float16)sf[j].z; p1.h[1] = (_Float16)sf[j].w;
                d[2 * j]     = p0.u;
                d[2 * j + 1] = p1.u;
            }
        } else {
            d[0] = sq[0].x; d[1] = sq[0].y; d[2] = sq[0].z; d[3] = sq[0].w;
            d[4] = sq[1].x; d[5] = sq[1].y; d[6] = sq[1].z; d[7] = sq[1].w;
        }
    };

    issueLoads(0);
    storeStage(0);
#pragma unroll
    for (int kt = 0; kt < KT; ++kt) {
        __syncthreads();                       // tile kt staged; prev reads done
        if (kt + 1 < KT) issueLoads(kt + 1);   // prefetch next tile (global)

        const uint32_t* sbuf = sA + (kt & 1) * SBUF;
        Frag a;
#pragma unroll
        for (int v = 0; v < 8; ++v) a.u[v] = sbuf[aBase + v];

#pragma unroll
        for (int nt = 0; nt < WN; ++nt) {
            Frag b;
            const uint4* wp =
                (const uint4*)Wp + ((size_t)(ntBase + nt) * KT + kt) * 64 + lane * 2;
            b.q[0] = wp[0];
            b.q[1] = wp[1];
            acc[nt] = __builtin_amdgcn_wmma_f32_16x16x32_f16(
                false, a.h, false, b.h, (short)0, acc[nt], false, false);
        }

        if (kt + 1 < KT) storeStage(kt + 1);   // write other LDS buffer
    }
}

template <int WN, int KT1, bool RELU, bool HAS2, bool SRC1F32>
__global__ __launch_bounds__(256, 1) void gemm_stage(
    const void* __restrict__ A1, const uint16_t* __restrict__ W1,
    const uint16_t* __restrict__ A2, const uint16_t* __restrict__ W2,
    const float* __restrict__ bias, float* __restrict__ out0,
    float* __restrict__ out1, uint16_t* __restrict__ outh, int ldo) {
    __shared__ uint32_t sA[2 * SBUF];
    const int blockRow = blockIdx.x * 128;
    const int ntBase   = blockIdx.y * WN;

    v8f acc[WN];
#pragma unroll
    for (int i = 0; i < WN; ++i)
#pragma unroll
        for (int r = 0; r < 8; ++r) acc[i][r] = 0.f;

    gemm_accum<WN, KT1, SRC1F32>(A1, W1, ntBase, acc, sA, blockRow);
    if constexpr (HAS2) {
        __syncthreads();   // protect LDS reuse between the two passes
        gemm_accum<WN, 16, false>(A2, W2, ntBase, acc, sA, blockRow);
    }

    const int tid = threadIdx.x, wave = tid >> 5, lane = tid & 31;
    const int m0   = blockRow + (wave << 4) + ((lane & 16) ? 8 : 0);
    const int ncol = lane & 15;
#pragma unroll
    for (int nt = 0; nt < WN; ++nt) {
        const int   col = (ntBase + nt) * 16 + ncol;
        const float bv  = bias[col];
#pragma unroll
        for (int r = 0; r < 8; ++r) {
            float v = acc[nt][r] + bv;
            if (RELU) v = v > 0.f ? v : 0.f;
            const size_t idx = (size_t)(m0 + r) * ldo + col;
            if (out0) out0[idx] = v;
            if (out1) out1[idx] = v;
            if (outh) {
                union { _Float16 h; uint16_t u; } cv;
                cv.h = (_Float16)v;
                outh[idx] = cv.u;
            }
        }
    }
}

// ---------------------------------------------------------------------------
extern "C" void kernel_launch(void* const* d_in, const int* in_sizes, int n_in,
                              void* d_out, int out_size, void* d_ws, size_t ws_size,
                              hipStream_t stream) {
    (void)in_sizes; (void)n_in; (void)out_size; (void)ws_size;

    auto F = [&](int i) { return (const float*)d_in[i]; };
    const float* sig = F(0);
    // FC params: 4 x (W, b) each
    const float* lW[4] = {F(1), F(3), F(5), F(7)};
    const float* lb[4] = {F(2), F(4), F(6), F(8)};
    const float* cW[4] = {F(9), F(11), F(13), F(15)};
    const float* cb[4] = {F(10), F(12), F(14), F(16)};
    const float* jW[4] = {F(17), F(19), F(21), F(23)};
    const float* jb[4] = {F(18), F(20), F(22), F(24)};
    const float* msgW = F(25); const float* msgb = F(26);
    const float* updW = F(27); const float* updb = F(28);

    // --- workspace layout -------------------------------------------------
    size_t off = 0;
    auto take = [&](size_t bytes) {
        size_t r = off;
        off = (off + bytes + 255) & ~(size_t)255;
        return r;
    };
    const size_t act512 = (size_t)B_ * 512 * 2;
    const size_t act160 = (size_t)B_ * HP * 2;
    const size_t w512x160 = (size_t)512 * HP * 2;
    const size_t w160x160 = (size_t)HP * HP * 2;
    const size_t w160x512 = (size_t)HP * 512 * 2;
    const size_t w512x512 = (size_t)512 * 512 * 2;

    size_t o_p1 = take(act512), o_X = take(act512), o_Y = take(act512);
    size_t o_hA = take(act160), o_hB = take(act160);
    size_t o_wj0t = take(w512x160), o_wj0b = take(w512x160);
    size_t o_wj1 = take(w160x160), o_wj2 = take(w160x160), o_wj3 = take(w160x512);
    size_t o_wl0t = take(w512x160), o_wl0b = take(w512x160), o_wl0c = take(w512x160);
    size_t o_wl1 = take(w160x160), o_wl2 = take(w160x160), o_wl3 = take(w160x512);
    size_t o_wc0c = take(w512x160);
    size_t o_wc1 = take(w160x160), o_wc2 = take(w160x160), o_wc3 = take(w160x512);
    size_t o_wm = take(w512x512), o_wu = take(w512x512);
    size_t o_bj0 = take(2048), o_bj1 = take(2048), o_bj2 = take(2048), o_bj3 = take(2048);
    size_t o_bl0 = take(2048), o_bl1 = take(2048), o_bl2 = take(2048), o_bl3 = take(2048);
    size_t o_bc0 = take(2048), o_bc1 = take(2048), o_bc2 = take(2048), o_bc3 = take(2048);
    size_t o_bm = take(2048), o_bu = take(2048);

    char* W = (char*)d_ws;
    auto WP = [&](size_t o) { return (uint16_t*)(W + o); };
    auto BP = [&](size_t o) { return (float*)(W + o); };
    auto AP = [&](size_t o) { return (const void*)(W + o); };

    float* outp = (float*)d_out;
    auto OUT = [&](int s) { return outp + (size_t)s * B_ * 512; };

    // --- pack weights / biases -------------------------------------------
    auto packW = [&](size_t dst, const float* s1, const float* s2, int ld,
                     int Kreal, int Nreal, int Kt, int Nt, float scale) {
        int total = Kt * Nt * 32;
        hipLaunchKernelGGL(pack_w_kernel, dim3((total + 255) / 256), dim3(256), 0,
                           stream, s1, s2, ld, Kreal, Nreal, Kt, Nt, WP(dst), scale);
    };
    auto packB = [&](size_t dst, const float* s, int Nreal, int Npad, float scale) {
        hipLaunchKernelGGL(pack_b_kernel, dim3((Npad + 255) / 256), dim3(256), 0,
                           stream, s, Nreal, Npad, BP(dst), scale);
    };

    // joint: Wj0 is (1024,132): top 512 rows = signal part, bottom = r part
    packW(o_wj0t, jW[0], nullptr, HR, 512, HR, 16, 10, 1.f);
    packW(o_wj0b, jW[0] + (size_t)512 * HR, nullptr, HR, 512, HR, 16, 10, 1.f);
    packW(o_wj1, jW[1], nullptr, HR, HR, HR, 5, 10, 1.f);
    packW(o_wj2, jW[2], nullptr, HR, HR, HR, 5, 10, 1.f);
    packW(o_wj3, jW[3], nullptr, 512, HR, 512, 5, 32, 1.f);
    // line: top / bottom / combined (cat(p1,p1) folds to top+bottom)
    packW(o_wl0t, lW[0], nullptr, HR, 512, HR, 16, 10, 1.f);
    packW(o_wl0b, lW[0] + (size_t)512 * HR, nullptr, HR, 512, HR, 16, 10, 1.f);
    packW(o_wl0c, lW[0], lW[0] + (size_t)512 * HR, HR, 512, HR, 16, 10, 1.f);
    packW(o_wl1, lW[1], nullptr, HR, HR, HR, 5, 10, 1.f);
    packW(o_wl2, lW[2], nullptr, HR, HR, HR, 5, 10, 1.f);
    packW(o_wl3, lW[3], nullptr, 512, HR, 512, 5, 32, 1.f);
    // circle combined
    packW(o_wc0c, cW[0], cW[0] + (size_t)512 * HR, HR, 512, HR, 16, 10, 1.f);
    packW(o_wc1, cW[1], nullptr, HR, HR, HR, 5, 10, 1.f);
    packW(o_wc2, cW[2], nullptr, HR, HR, HR, 5, 10, 1.f);
    packW(o_wc3, cW[3], nullptr, 512, HR, 512, 5, 32, 1.f);
    // msg (x2: c1+c2 == 2*c1), upd
    packW(o_wm, msgW, nullptr, 512, 512, 512, 16, 32, 2.f);
    packW(o_wu, updW, nullptr, 512, 512, 512, 16, 32, 1.f);

    packB(o_bj0, jb[0], HR, HP, 1.f); packB(o_bj1, jb[1], HR, HP, 1.f);
    packB(o_bj2, jb[2], HR, HP, 1.f); packB(o_bj3, jb[3], 512, 512, 1.f);
    packB(o_bl0, lb[0], HR, HP, 1.f); packB(o_bl1, lb[1], HR, HP, 1.f);
    packB(o_bl2, lb[2], HR, HP, 1.f); packB(o_bl3, lb[3], 512, 512, 1.f);
    packB(o_bc0, cb[0], HR, HP, 1.f); packB(o_bc1, cb[1], HR, HP, 1.f);
    packB(o_bc2, cb[2], HR, HP, 1.f); packB(o_bc3, cb[3], 512, 512, 1.f);
    packB(o_bm, msgb, 512, 512, 2.f); packB(o_bu, updb, 512, 512, 1.f);

    const dim3 g160(NBM, 1), g512(NBM, 4), blk(256);

    // --- p1 = joint(cat(signal, 0)) --------------------------------------
    gemm_stage<10, 16, true, false, true><<<g160, blk, 0, stream>>>(
        sig, WP(o_wj0t), nullptr, nullptr, BP(o_bj0),
        nullptr, nullptr, WP(o_hA), HP);
    gemm_stage<10, 5, true, false, false><<<g160, blk, 0, stream>>>(
        AP(o_hA), WP(o_wj1), nullptr, nullptr, BP(o_bj1),
        nullptr, nullptr, WP(o_hB), HP);
    gemm_stage<10, 5, true, false, false><<<g160, blk, 0, stream>>>(
        AP(o_hB), WP(o_wj2), nullptr, nullptr, BP(o_bj2),
        nullptr, nullptr, WP(o_hA), HP);
    gemm_stage<8, 5, false, false, false><<<g512, blk, 0, stream>>>(
        AP(o_hA), WP(o_wj3), nullptr, nullptr, BP(o_bj3),
        OUT(0), OUT(1), WP(o_p1), 512);

    // --- l1 = line(cat(p1,p1)) via combined weights -----------------------
    gemm_stage<10, 16, true, false, false><<<g160, blk, 0, stream>>>(
        AP(o_p1), WP(o_wl0c), nullptr, nullptr, BP(o_bl0),
        nullptr, nullptr, WP(o_hA), HP);
    gemm_stage<10, 5, true, false, false><<<g160, blk, 0, stream>>>(
        AP(o_hA), WP(o_wl1), nullptr, nullptr, BP(o_bl1),
        nullptr, nullptr, WP(o_hB), HP);
    gemm_stage<10, 5, true, false, false><<<g160, blk, 0, stream>>>(
        AP(o_hB), WP(o_wl2), nullptr, nullptr, BP(o_bl2),
        nullptr, nullptr, WP(o_hA), HP);
    gemm_stage<8, 5, false, false, false><<<g512, blk, 0, stream>>>(
        AP(o_hA), WP(o_wl3), nullptr, nullptr, BP(o_bl3),
        OUT(2), nullptr, nullptr, 512);

    // --- c1 = circle(cat(p1,p1)); c2 == c1 --------------------------------
    gemm_stage<10, 16, true, false, false><<<g160, blk, 0, stream>>>(
        AP(o_p1), WP(o_wc0c), nullptr, nullptr, BP(o_bc0),
        nullptr, nullptr, WP(o_hA), HP);
    gemm_stage<10, 5, true, false, false><<<g160, blk, 0, stream>>>(
        AP(o_hA), WP(o_wc1), nullptr, nullptr, BP(o_bc1),
        nullptr, nullptr, WP(o_hB), HP);
    gemm_stage<10, 5, true, false, false><<<g160, blk, 0, stream>>>(
        AP(o_hB), WP(o_wc2), nullptr, nullptr, BP(o_bc2),
        nullptr, nullptr, WP(o_hA), HP);
    gemm_stage<8, 5, false, false, false><<<g512, blk, 0, stream>>>(
        AP(o_hA), WP(o_wc3), nullptr, nullptr, BP(o_bc3),
        OUT(3), OUT(4), WP(o_X), 512);

    // --- m = 2*(c1@Mw + mb);  r = m@Uw + ub --------------------------------
    gemm_stage<8, 16, false, false, false><<<g512, blk, 0, stream>>>(
        AP(o_X), WP(o_wm), nullptr, nullptr, BP(o_bm),
        nullptr, nullptr, WP(o_Y), 512);
    gemm_stage<8, 16, false, false, false><<<g512, blk, 0, stream>>>(
        AP(o_Y), WP(o_wu), nullptr, nullptr, BP(o_bu),
        nullptr, nullptr, WP(o_X), 512);

    // --- p3 = joint(cat(signal, r)) ---------------------------------------
    gemm_stage<10, 16, true, true, true><<<g160, blk, 0, stream>>>(
        sig, WP(o_wj0t), WP(o_X), WP(o_wj0b), BP(o_bj0),
        nullptr, nullptr, WP(o_hA), HP);
    gemm_stage<10, 5, true, false, false><<<g160, blk, 0, stream>>>(
        AP(o_hA), WP(o_wj1), nullptr, nullptr, BP(o_bj1),
        nullptr, nullptr, WP(o_hB), HP);
    gemm_stage<10, 5, true, false, false><<<g160, blk, 0, stream>>>(
        AP(o_hB), WP(o_wj2), nullptr, nullptr, BP(o_bj2),
        nullptr, nullptr, WP(o_hA), HP);
    gemm_stage<8, 5, false, false, false><<<g512, blk, 0, stream>>>(
        AP(o_hA), WP(o_wj3), nullptr, nullptr, BP(o_bj3),
        OUT(5), nullptr, WP(o_Y), 512);

    // --- l2 = line(cat(p1, p3)); l3 == l2 ---------------------------------
    gemm_stage<10, 16, true, true, false><<<g160, blk, 0, stream>>>(
        AP(o_p1), WP(o_wl0t), WP(o_Y), WP(o_wl0b), BP(o_bl0),
        nullptr, nullptr, WP(o_hA), HP);
    gemm_stage<10, 5, true, false, false><<<g160, blk, 0, stream>>>(
        AP(o_hA), WP(o_wl1), nullptr, nullptr, BP(o_bl1),
        nullptr, nullptr, WP(o_hB), HP);
    gemm_stage<10, 5, true, false, false><<<g160, blk, 0, stream>>>(
        AP(o_hB), WP(o_wl2), nullptr, nullptr, BP(o_bl2),
        nullptr, nullptr, WP(o_hA), HP);
    gemm_stage<8, 5, false, false, false><<<g512, blk, 0, stream>>>(
        AP(o_hA), WP(o_wl3), nullptr, nullptr, BP(o_bl3),
        OUT(6), OUT(7), nullptr, 512);
}
